// clique_MPNN_46256797778564
// MI455X (gfx1250) — matrix operand order repaired and checked by reading, not compiled, not run on hardware
//
#include <hip/hip_runtime.h>
#include <hip/hip_bf16.h>
#include <stdint.h>

#define TPB 256
#define F 128            // hidden width
#define BN_EPS 1e-5f
#define SLOPE 0.01f

typedef _Float16 v16h __attribute__((ext_vector_type(16)));
typedef _Float16 v8h  __attribute__((ext_vector_type(8)));
typedef float    v8f  __attribute__((ext_vector_type(8)));

__device__ __forceinline__ float lrelu_f(float v) { return v > 0.f ? v : SLOPE * v; }

// ---------------------------------------------------------------------------
// WMMA GEMM: C[N x NT*16] = act(A[N x 128] @ B[128 x NT*16] + bias) (* mask)
// A row-major f16; BT is B transposed (col-major): BT[n*128 + k], f16.
// B staged into LDS once per block (stride padded to 136 halves to avoid bank
// conflicts). Each of 8 waves computes a 16-row strip x NT*16 cols, K=128.
// Per K-block, ALL NT B fragments are loaded into distinct registers before
// the WMMA chain, so ds_load latency overlaps matrix-core execution.
// ---------------------------------------------------------------------------
template<int NT, int ACT, bool OUTF, bool OUTH, bool MUL_MASK>
__global__ void __launch_bounds__(256) gemm_wmma(
    const _Float16* __restrict__ A,
    const _Float16* __restrict__ BT,
    const float* __restrict__ bias,
    const int* __restrict__ maskInt,
    float* __restrict__ Cf,
    _Float16* __restrict__ Ch,
    int nrows)
{
  constexpr int NCOL = NT * 16;
  constexpr int LSTR = 136;                 // padded row stride in halves (272B)
  __shared__ _Float16 sBT[NCOL * LSTR];

  const int tid = threadIdx.x;

  // ---- cooperative stage: BT (NCOL x 128 halves) -> LDS, 16B chunks ----
  for (int c = tid; c < NCOL * 16; c += 256) {
    int n = c >> 4, seg = c & 15;
    *(v8h*)(&sBT[n * LSTR + seg * 8]) = *(const v8h*)(BT + (size_t)n * 128 + seg * 8);
  }
  __syncthreads();

  const int lane = tid & 31;
  const int wave = blockIdx.x * 8 + (tid >> 5);
  const int m0 = wave * 16;
  if (m0 >= nrows) return;           // wave-uniform: EXEC all-ones below
  const int half = lane >> 4;        // 0 or 1
  const int l16  = lane & 15;

  // ---- preload all 4 A fragments (one clause of global b128 loads) ----
  // 16-bit A layout: lanes 0-15 hold row M=l16, K = kb+0..7 (elems 0-7) and
  // kb+16..23 (elems 8-15); lanes 16-31 hold K = kb+8..15 / kb+24..31.
  v16h a[4];
#pragma unroll
  for (int kk = 0; kk < 4; ++kk) {
    const _Float16* ap = A + (size_t)(m0 + l16) * 128 + kk * 32 + half * 8;
    v8h lo = *(const v8h*)(ap);
    v8h hi = *(const v8h*)(ap + 16);
#pragma unroll
    for (int i = 0; i < 8; ++i) { a[kk][i] = lo[i]; a[kk][8 + i] = hi[i]; }
  }

  // B fragment loader: lane holds column N=l16 of tile t;
  // lanes 0-15: K=kb+0..7 / kb+16..23, lanes 16-31: K=kb+8..15 / kb+24..31.
  auto loadB = [&](int kk, int t) -> v16h {
    const _Float16* bp = &sBT[(t * 16 + l16) * LSTR + kk * 32 + half * 8];
    v8h lo = *(const v8h*)(bp);
    v8h hi = *(const v8h*)(bp + 16);
    v16h b;
#pragma unroll
    for (int i = 0; i < 8; ++i) { b[i] = lo[i]; b[8 + i] = hi[i]; }
    return b;
  };

  v8f zero = {};
  v8f acc[NT];
#pragma unroll
  for (int t = 0; t < NT; ++t) acc[t] = zero;

  // ---- K loop: load ALL NT B fragments (distinct regs), then WMMA chain ----
#pragma unroll
  for (int kk = 0; kk < 4; ++kk) {
    v16h bb[NT];
#pragma unroll
    for (int t = 0; t < NT; ++t) bb[t] = loadB(kk, t);
#pragma unroll
    for (int t = 0; t < NT; ++t) {
      acc[t] = __builtin_amdgcn_wmma_f32_16x16x32_f16(
          false, a[kk], false, bb[t], (short)0, acc[t], false, false);
    }
  }

  // Epilogue. C/D layout: VGPR v, lanes 0-15 -> (M = v,   N = l16)
  //                               lanes 16-31 -> (M = v+8, N = l16)
#pragma unroll
  for (int t = 0; t < NT; ++t) {
#pragma unroll
    for (int v = 0; v < 8; ++v) {
      int row = m0 + v + half * 8;
      int col = t * 16 + l16;
      float val = acc[t][v] + bias[col];
      if (ACT == 1) val = val > 0.f ? val : 0.f;
      else if (ACT == 2) val = lrelu_f(val);
      if (MUL_MASK) val *= (float)maskInt[row];
      if (OUTF) Cf[(size_t)row * NCOL + col] = val;
      if (OUTH) Ch[(size_t)row * NCOL + col] = (_Float16)val;
    }
  }
}

// ---------------------------------------------------------------------------
// Elementwise / scatter kernels
// ---------------------------------------------------------------------------
__global__ void cnt_kernel(const int* __restrict__ batch, float* cnt, int n) {
  int i = blockIdx.x * blockDim.x + threadIdx.x;
  if (i < n) atomicAdd(&cnt[batch[i]], 1.0f);
}

__global__ void mask_init(const float* __restrict__ x, int* m, int n) {
  int i = blockIdx.x * blockDim.x + threadIdx.x;
  if (i < n) m[i] = (fabsf(x[i]) > 0.f) ? 1 : 0;
}

__global__ void mask_copy(const int* __restrict__ src, int* dst, int n) {
  int i = blockIdx.x * blockDim.x + threadIdx.x;
  if (i < n) dst[i] = src[i];
}

__global__ void mask_edge(const int* __restrict__ row, const int* __restrict__ col,
                          const int* __restrict__ mcur, int* mnext, int e) {
  int i = blockIdx.x * blockDim.x + threadIdx.x;
  if (i < e) atomicMax(&mnext[col[i]], mcur[row[i]]);
}

__global__ void edge_scatter1(const int* __restrict__ row, const int* __restrict__ col,
                              const float* __restrict__ x, float* agg, int e) {
  int i = blockIdx.x * blockDim.x + threadIdx.x;
  if (i < e) atomicAdd(&agg[col[i]], x[row[i]]);
}

__global__ void edge_scatter128(const int* __restrict__ row, const int* __restrict__ col,
                                const float* __restrict__ H, float* __restrict__ AGG, int e) {
  unsigned gid = blockIdx.x * blockDim.x + threadIdx.x;
  int ei = gid >> 5;               // 32 lanes per edge, 4 floats each
  int q  = gid & 31;
  if (ei >= e) return;
  int r = row[ei], c = col[ei];
  const float4 hv = *(const float4*)(H + (size_t)r * F + q * 4);
  float* dst = AGG + (size_t)c * F + q * 4;
  atomicAdd(dst + 0, hv.x);
  atomicAdd(dst + 1, hv.y);
  atomicAdd(dst + 2, hv.z);
  atomicAdd(dst + 3, hv.w);
}

// z1 = relu(((1+eps)*x + agg1) * w1 + b1) -> f16
__global__ void conv1_z1(const float* __restrict__ x, const float* __restrict__ agg1,
                         const float* __restrict__ epsp, const float* __restrict__ w1,
                         const float* __restrict__ b1, _Float16* __restrict__ Z16,
                         size_t total) {
  size_t idx = (size_t)blockIdx.x * blockDim.x + threadIdx.x;
  if (idx >= total) return;
  size_t r = idx >> 7;
  int c = idx & 127;
  float z0 = (1.f + epsp[0]) * x[r] + agg1[r];
  float v = z0 * w1[c] + b1[c];
  Z16[idx] = (_Float16)(v > 0.f ? v : 0.f);
}

// Z16 = (f16)((1+eps)*H + AGG)
__global__ void z_combine(const float* __restrict__ H, const float* __restrict__ AGG,
                          const float* __restrict__ epsp, _Float16* __restrict__ Z16,
                          size_t total) {
  size_t idx = (size_t)blockIdx.x * blockDim.x + threadIdx.x;
  if (idx >= total) return;
  Z16[idx] = (_Float16)((1.f + epsp[0]) * H[idx] + AGG[idx]);
}

// Per-feature sum / sumsq with LDS partials
__global__ void __launch_bounds__(256) bn_stats(const float* __restrict__ X,
                                                float* sum, float* sumsq, int nrows) {
  __shared__ float ss[F], sq[F];
  int t = threadIdx.x;
  if (t < F) { ss[t] = 0.f; sq[t] = 0.f; }
  __syncthreads();
  int col = t & (F - 1);
  float ls = 0.f, lq = 0.f;
  for (int r = blockIdx.x * 2 + (t >> 7); r < nrows; r += gridDim.x * 2) {
    float v = X[(size_t)r * F + col];
    ls += v; lq += v * v;
  }
  atomicAdd(&ss[col], ls);
  atomicAdd(&sq[col], lq);
  __syncthreads();
  if (t < F) { atomicAdd(&sum[t], ss[t]); atomicAdd(&sumsq[t], sq[t]); }
}

__global__ void bn_finalize(const float* __restrict__ sum, const float* __restrict__ sumsq,
                            const float* __restrict__ g, const float* __restrict__ b,
                            float* scale, float* shift, int nrows) {
  int t = threadIdx.x;
  if (t >= F) return;
  float inv = 1.f / (float)nrows;
  float mean = sum[t] * inv;
  float var = sumsq[t] * inv - mean * mean;
  float sc = g[t] * rsqrtf(var + BN_EPS);
  scale[t] = sc;
  shift[t] = b[t] - mean * sc;
}

// U = (optional H residual + lrelu(T2*scale+shift)) * mask * gsn   (in place on T2)
template<bool RESID>
__global__ void apply_pre(float* __restrict__ UA, const float* __restrict__ H,
                          const float* __restrict__ scale, const float* __restrict__ shift,
                          const int* __restrict__ maskInt, const int* __restrict__ batch,
                          const float* __restrict__ cnt, size_t total) {
  size_t idx = (size_t)blockIdx.x * blockDim.x + threadIdx.x;
  if (idx >= total) return;
  size_t r = idx >> 7;
  int c = idx & 127;
  float v = lrelu_f(UA[idx] * scale[c] + shift[c]);
  if (RESID) v = H[idx] + v;
  float gsn = rsqrtf(cnt[batch[r]]);
  UA[idx] = v * (float)maskInt[r] * gsn;
}

// H = U*scale+shift; also write f16 copy
__global__ void apply_post(const float* __restrict__ U, const float* __restrict__ scale,
                           const float* __restrict__ shift, float* __restrict__ H,
                           _Float16* __restrict__ H16, size_t total) {
  size_t idx = (size_t)blockIdx.x * blockDim.x + threadIdx.x;
  if (idx >= total) return;
  int c = idx & 127;
  float v = U[idx] * scale[c] + shift[c];
  H[idx] = v;
  H16[idx] = (_Float16)v;
}

// transpose+convert: W[K x Ncols] row-major -> WT[Ncols x K] f16
__global__ void wtrans(const float* __restrict__ W, _Float16* __restrict__ WT,
                       int K, int Ncols) {
  int idx = blockIdx.x * blockDim.x + threadIdx.x;
  if (idx >= K * Ncols) return;
  int n = idx / K, k = idx % K;
  WT[(size_t)n * K + k] = (_Float16)W[(size_t)k * Ncols + n];
}

__global__ void lin2_k(const float* __restrict__ HS, const float* __restrict__ w,
                       const float* __restrict__ b, const int* __restrict__ maskInt,
                       float* __restrict__ HF, int n) {
  int i = blockIdx.x * blockDim.x + threadIdx.x;
  if (i >= n) return;
  float acc = b[0];
#pragma unroll
  for (int j = 0; j < 16; ++j) acc += HS[(size_t)i * 16 + j] * w[j];
  HF[i] = lrelu_f(acc) * (float)maskInt[i];
}

// per-graph max/min -> probs -> sum & sumsq (nodes contiguous per graph)
__global__ void __launch_bounds__(256) graph_reduce(const float* __restrict__ HF,
                                                    float* __restrict__ probs,
                                                    float* S, float* SELF, int npg) {
  __shared__ float sa[256], sb[256];
  int g = blockIdx.x, t = threadIdx.x;
  const float* hf = HF + (size_t)g * npg;
  float mx = -3.4e38f, mn = 3.4e38f;
  for (int i = t; i < npg; i += 256) {
    float v = hf[i];
    mx = fmaxf(mx, v); mn = fminf(mn, v);
  }
  sa[t] = mx; sb[t] = mn;
  __syncthreads();
  for (int s = 128; s > 0; s >>= 1) {
    if (t < s) { sa[t] = fmaxf(sa[t], sa[t + s]); sb[t] = fminf(sb[t], sb[t + s]); }
    __syncthreads();
  }
  float bmax = sa[0], bmin = sb[0];
  __syncthreads();
  float inv = 1.f / (bmax + 1e-6f - bmin);
  float ls = 0.f, lq = 0.f;
  for (int i = t; i < npg; i += 256) {
    float p = (hf[i] - bmin) * inv;
    probs[(size_t)g * npg + i] = p;
    ls += p; lq += p * p;
  }
  sa[t] = ls; sb[t] = lq;
  __syncthreads();
  for (int s = 128; s > 0; s >>= 1) {
    if (t < s) { sa[t] += sa[t + s]; sb[t] += sb[t + s]; }
    __syncthreads();
  }
  if (t == 0) { S[g] = sa[0]; SELF[g] = sb[0]; }
}

__global__ void edge_w(const int* __restrict__ row, const int* __restrict__ col,
                       const float* __restrict__ weights, const float* __restrict__ probs,
                       const int* __restrict__ batch, float* EW, int e) {
  int i = blockIdx.x * blockDim.x + threadIdx.x;
  if (i >= e) return;
  int r = row[i], c = col[i];
  if (r == c) return;
  atomicAdd(&EW[batch[r]], weights[r] * probs[r] * probs[c]);
}

__global__ void loss_k(const float* __restrict__ S, const float* __restrict__ SELF,
                       const float* __restrict__ EW, float* out, int g) {
  int i = blockIdx.x * blockDim.x + threadIdx.x;
  if (i >= g) return;
  float ewG = EW[i] * 0.5f;
  float ecw = 0.5f * S[i] * S[i] - SELF[i];
  out[i] = 0.125f * (ecw - ewG) - 0.5f * ewG;
}

// ---------------------------------------------------------------------------
extern "C" void kernel_launch(void* const* d_in, const int* in_sizes, int n_in,
                              void* d_out, int out_size, void* d_ws, size_t ws_size,
                              hipStream_t stream) {
  (void)n_in; (void)ws_size;
  const float* x       = (const float*)d_in[0];
  const float* weights = (const float*)d_in[1];
  const int*   eidx    = (const int*)d_in[2];
  const int*   batch   = (const int*)d_in[3];
  const float* c1_w1   = (const float*)d_in[4];
  const float* c1_b1   = (const float*)d_in[5];
  const float* c1_w2   = (const float*)d_in[6];
  const float* c1_b2   = (const float*)d_in[7];
  const float* c1_bng  = (const float*)d_in[8];
  const float* c1_bnb  = (const float*)d_in[9];
  const float* c1_eps  = (const float*)d_in[10];
  const float* cs_w1   = (const float*)d_in[11];
  const float* cs_b1   = (const float*)d_in[12];
  const float* cs_w2   = (const float*)d_in[13];
  const float* cs_b2   = (const float*)d_in[14];
  const float* cs_bng  = (const float*)d_in[15];
  const float* cs_bnb  = (const float*)d_in[16];
  const float* cs_eps  = (const float*)d_in[17];
  const float* bn1_g   = (const float*)d_in[18];
  const float* bn1_b   = (const float*)d_in[19];
  const float* bns_g   = (const float*)d_in[20];
  const float* bns_b   = (const float*)d_in[21];
  const float* lin1_w  = (const float*)d_in[22];
  const float* lin1_b  = (const float*)d_in[23];
  const float* lin2_w  = (const float*)d_in[24];
  const float* lin2_b  = (const float*)d_in[25];

  const int N = in_sizes[0];
  const int E = in_sizes[2] / 2;
  const int G = out_size;
  const int NPG = N / G;
  const int* erow = eidx;
  const int* ecol = eidx + E;
  const size_t NF = (size_t)N * F;

  // workspace carve-up
  char* ws = (char*)d_ws;
  size_t off = 0;
  auto take = [&](size_t bytes) -> char* {
    char* p = ws + off;
    off = (off + bytes + 255) & ~(size_t)255;
    return p;
  };
  float*    H     = (float*)take(NF * 4);
  float*    AGG   = (float*)take(NF * 4);     // agg / T2 / U (reused)
  _Float16* F16A  = (_Float16*)take(NF * 2);  // z16 / h16
  _Float16* F16B  = (_Float16*)take(NF * 2);  // t16
  _Float16* c1w2t = (_Float16*)take(F * F * 2);
  _Float16* w1t0  = (_Float16*)take(F * F * 2);
  _Float16* w2t0  = (_Float16*)take(F * F * 2);
  _Float16* w1t1  = (_Float16*)take(F * F * 2);
  _Float16* w2t1  = (_Float16*)take(F * F * 2);
  _Float16* lin1t = (_Float16*)take(16 * F * 2);
  float*    agg1  = (float*)take((size_t)N * 4);
  int*      maskA = (int*)take((size_t)N * 4);
  int*      maskB = (int*)take((size_t)N * 4);
  float*    cnt   = (float*)take((size_t)G * 4);
  float*    sum   = (float*)take(F * 4);
  float*    sumsq = (float*)take(F * 4);
  float*    scale = (float*)take(F * 4);
  float*    shift = (float*)take(F * 4);
  float*    HS    = (float*)take((size_t)N * 16 * 4);
  float*    HF    = (float*)take((size_t)N * 4);
  float*    probs = (float*)take((size_t)N * 4);
  float*    Sg    = (float*)take((size_t)G * 4);
  float*    SELF  = (float*)take((size_t)G * 4);
  float*    EW    = (float*)take((size_t)G * 4);

  auto b1d = [](long long n) { return (int)((n + TPB - 1) / TPB); };
  const int gemm_blocks = (int)(((N + 15) / 16 + 7) / 8);  // 8 waves/block

  auto bnpair = [&](const float* X, const float* g, const float* b) {
    hipMemsetAsync(sum, 0, F * 4, stream);
    hipMemsetAsync(sumsq, 0, F * 4, stream);
    bn_stats<<<512, TPB, 0, stream>>>(X, sum, sumsq, N);
    bn_finalize<<<1, 128, 0, stream>>>(sum, sumsq, g, b, scale, shift, N);
  };

  // ---- prologue: counts, initial mask, weight transposes ----
  hipMemsetAsync(cnt, 0, G * 4, stream);
  cnt_kernel<<<b1d(N), TPB, 0, stream>>>(batch, cnt, N);
  mask_init<<<b1d(N), TPB, 0, stream>>>(x, maskA, N);
  mask_copy<<<b1d(N), TPB, 0, stream>>>(maskA, maskB, N);
  mask_edge<<<b1d(E), TPB, 0, stream>>>(erow, ecol, maskA, maskB, E);
  wtrans<<<b1d(F * F), TPB, 0, stream>>>(c1_w2, c1w2t, F, F);
  wtrans<<<b1d(F * F), TPB, 0, stream>>>(cs_w1, w1t0, F, F);
  wtrans<<<b1d(F * F), TPB, 0, stream>>>(cs_w2, w2t0, F, F);
  wtrans<<<b1d(F * F), TPB, 0, stream>>>(cs_w1 + F * F, w1t1, F, F);
  wtrans<<<b1d(F * F), TPB, 0, stream>>>(cs_w2 + F * F, w2t1, F, F);
  wtrans<<<b1d(16 * F), TPB, 0, stream>>>(lin1_w, lin1t, F, 16);

  // ---- conv1 ----
  hipMemsetAsync(agg1, 0, (size_t)N * 4, stream);
  edge_scatter1<<<b1d(E), TPB, 0, stream>>>(erow, ecol, x, agg1, E);
  conv1_z1<<<b1d(NF), TPB, 0, stream>>>(x, agg1, c1_eps, c1_w1, c1_b1, F16A, NF);
  gemm_wmma<8, 1, true, false, false><<<gemm_blocks, 256, 0, stream>>>(
      F16A, c1w2t, c1_b2, nullptr, AGG, nullptr, N);
  bnpair(AGG, c1_bng, c1_bnb);
  apply_pre<false><<<b1d(NF), TPB, 0, stream>>>(AGG, nullptr, scale, shift,
                                                maskB, batch, cnt, NF);
  bnpair(AGG, bn1_g, bn1_b);
  apply_post<<<b1d(NF), TPB, 0, stream>>>(AGG, scale, shift, H, F16A, NF);

  // ---- GIN layers ----
  int* mcur = maskB;
  int* mnext = maskA;
  for (int i = 0; i < 2; ++i) {
    hipMemsetAsync(AGG, 0, NF * 4, stream);
    edge_scatter128<<<b1d((long long)E * 32), TPB, 0, stream>>>(erow, ecol, H, AGG, E);
    z_combine<<<b1d(NF), TPB, 0, stream>>>(H, AGG, cs_eps + i, F16A, NF);
    gemm_wmma<8, 1, false, true, false><<<gemm_blocks, 256, 0, stream>>>(
        F16A, (i == 0 ? w1t0 : w1t1), cs_b1 + i * F, nullptr, nullptr, F16B, N);
    gemm_wmma<8, 1, true, false, false><<<gemm_blocks, 256, 0, stream>>>(
        F16B, (i == 0 ? w2t0 : w2t1), cs_b2 + i * F, nullptr, AGG, nullptr, N);
    bnpair(AGG, cs_bng + i * F, cs_bnb + i * F);
    mask_copy<<<b1d(N), TPB, 0, stream>>>(mcur, mnext, N);
    mask_edge<<<b1d(E), TPB, 0, stream>>>(erow, ecol, mcur, mnext, E);
    { int* t = mcur; mcur = mnext; mnext = t; }
    apply_pre<true><<<b1d(NF), TPB, 0, stream>>>(AGG, H, scale, shift,
                                                 mcur, batch, cnt, NF);
    bnpair(AGG, bns_g + i * F, bns_b + i * F);
    apply_post<<<b1d(NF), TPB, 0, stream>>>(AGG, scale, shift, H, F16A, NF);
  }

  // ---- head ----
  gemm_wmma<1, 2, true, false, true><<<gemm_blocks, 256, 0, stream>>>(
      F16A, lin1t, lin1_b, mcur, HS, nullptr, N);
  lin2_k<<<b1d(N), TPB, 0, stream>>>(HS, lin2_w, lin2_b, mcur, HF, N);
  graph_reduce<<<G, 256, 0, stream>>>(HF, probs, Sg, SELF, NPG);
  hipMemsetAsync(EW, 0, (size_t)G * 4, stream);
  edge_w<<<b1d(E), TPB, 0, stream>>>(erow, ecol, weights, probs, batch, EW, E);
  loss_k<<<1, 128, 0, stream>>>(Sg, SELF, EW, (float*)d_out, G);
}
